// Attention_84911503442315
// MI455X (gfx1250) — matrix operand order repaired and checked by reading, not compiled, not run on hardware
//
#include <hip/hip_runtime.h>
#include <hip/hip_bf16.h>

typedef unsigned short u16b;
typedef __attribute__((ext_vector_type(16))) __bf16 v16bf;
typedef __attribute__((ext_vector_type(8)))  float  v8f;

struct alignas(32) Frag16 { uint4 lo, hi; };

static __device__ __forceinline__ v16bf frag_as(const Frag16& f) {
  v16bf r; __builtin_memcpy(&r, &f, sizeof(r)); return r;
}
static __device__ __forceinline__ u16b to_bf16_bits(float x) {
  unsigned u = __float_as_uint(x);
  u += 0x7FFFu + ((u >> 16) & 1u);        // round-to-nearest-even
  return (u16b)(u >> 16);
}

// ---------------------------------------------------------------- convert
__global__ void cvt_f32_to_bf16(const float* __restrict__ in,
                                u16b* __restrict__ out, int n4) {
  int i = blockIdx.x * blockDim.x + threadIdx.x;
  if (i >= n4) return;
  float4 v = ((const float4*)in)[i];
  uint2 p;
  p.x = (unsigned)to_bf16_bits(v.x) | ((unsigned)to_bf16_bits(v.y) << 16);
  p.y = (unsigned)to_bf16_bits(v.z) | ((unsigned)to_bf16_bits(v.w) << 16);
  ((uint2*)out)[i] = p;
}

// ---------------------------------------------------------------- GEMM core
// C[M,N] = A[M,K] (bf16 rm) x B[K,N] (bf16 rm). Block 128x128, K-step 64,
// 256 threads = 8 waves, wave tile 32x64 (2x4 WMMA tiles), 16 WMMA / K-step.
#define GBM 128
#define GBN 128
#define GBK 64

// Shared compute body: accumulates the 2x4 accumulator grid for this wave.
static __device__ __forceinline__ void gemm_tiles(
    const u16b* __restrict__ A, const u16b* __restrict__ B,
    int N, int K, int m0, int n0,
    u16b* As, u16b* Bst,
    int tid, int wm, int wn, int lrow, int kb8, int kb16,
    v8f acc[2][4])
{
  const int arow = tid >> 1, acb = (tid & 1) * 32;   // A: 128 rows x 64 cols
  const int brow = tid >> 2, bnb = (tid & 3) * 32;   // B: 64 rows x 128 cols

  for (int k0 = 0; k0 < K; k0 += GBK) {
    __syncthreads();
    { // A tile -> LDS row-major
      const uint4* s4 = (const uint4*)(A + (size_t)(m0 + arow) * K + k0 + acb);
      uint4* d4 = (uint4*)(As + arow * GBK + acb);
      d4[0] = s4[0]; d4[1] = s4[1]; d4[2] = s4[2]; d4[3] = s4[3];
    }
    { // B tile -> LDS transposed: Bst[n][k]
      const uint4* s4 = (const uint4*)(B + (size_t)(k0 + brow) * N + n0 + bnb);
      #pragma unroll
      for (int w = 0; w < 4; ++w) {
        uint4 q = s4[w];
        unsigned vals[4] = {q.x, q.y, q.z, q.w};
        #pragma unroll
        for (int c = 0; c < 4; ++c) {
          int j = bnb + w * 8 + c * 2;
          Bst[(j    ) * GBK + brow] = (u16b)(vals[c] & 0xffffu);
          Bst[(j + 1) * GBK + brow] = (u16b)(vals[c] >> 16);
        }
      }
    }
    __syncthreads();

    #pragma unroll
    for (int kc = 0; kc < 2; ++kc) {
      v16bf afrag[2];
      #pragma unroll
      for (int mt = 0; mt < 2; ++mt) {
        const uint4* p = (const uint4*)(As + (wm * 32 + mt * 16 + lrow) * GBK + kc * 32);
        Frag16 f; f.lo = p[kb8 >> 3]; f.hi = p[(16 + kb8) >> 3];
        afrag[mt] = frag_as(f);
      }
      #pragma unroll
      for (int nt = 0; nt < 4; ++nt) {
        const uint4* p = (const uint4*)(Bst + (wn * 64 + nt * 16 + lrow) * GBK + kc * 32);
        Frag16 f; f.lo = p[kb8 >> 3]; f.hi = p[(16 + kb8) >> 3];
        v16bf bfrag = frag_as(f);
        #pragma unroll
        for (int mt = 0; mt < 2; ++mt)
          acc[mt][nt] = __builtin_amdgcn_wmma_f32_16x16x32_bf16(
              false, afrag[mt], false, bfrag, (short)0, acc[mt][nt], false, false);
      }
    }
  }
}

// Variant 1: bf16 output, no bias (QKV GEMM).
__global__ __launch_bounds__(256) void gemm_bf16_out(
    const u16b* __restrict__ A, const u16b* __restrict__ B,
    int M, int N, int K, u16b* __restrict__ C)
{
  __shared__ alignas(16) u16b As [GBM * GBK];
  __shared__ alignas(16) u16b Bst[GBN * GBK];
  const int tid = threadIdx.x, wave = tid >> 5, lane = tid & 31;
  const int lrow = lane & 15, lhalf = lane >> 4;
  const int kb8 = lhalf * 8, kb16 = lhalf * 16;
  const int wm = wave & 3, wn = wave >> 2;
  const int m0 = blockIdx.y * GBM, n0 = blockIdx.x * GBN;

  v8f acc[2][4];
  #pragma unroll
  for (int i = 0; i < 2; ++i)
    #pragma unroll
    for (int j = 0; j < 4; ++j)
      #pragma unroll
      for (int e = 0; e < 8; ++e) acc[i][j][e] = 0.0f;

  gemm_tiles(A, B, N, K, m0, n0, As, Bst, tid, wm, wn, lrow, kb8, kb16, acc);

  #pragma unroll
  for (int mt = 0; mt < 2; ++mt)
    #pragma unroll
    for (int nt = 0; nt < 4; ++nt) {
      int c = n0 + wn * 64 + nt * 16 + lrow;
      #pragma unroll
      for (int v = 0; v < 8; ++v) {
        int r = m0 + wm * 32 + mt * 16 + v + lhalf * 8;
        C[(size_t)r * N + c] = to_bf16_bits(acc[mt][nt][v]);
      }
    }
}

// Variant 2: f32 output + bias (projection GEMM).
__global__ __launch_bounds__(256) void gemm_f32_bias_out(
    const u16b* __restrict__ A, const u16b* __restrict__ B,
    int M, int N, int K, float* __restrict__ C,
    const float* __restrict__ bias)
{
  __shared__ alignas(16) u16b As [GBM * GBK];
  __shared__ alignas(16) u16b Bst[GBN * GBK];
  const int tid = threadIdx.x, wave = tid >> 5, lane = tid & 31;
  const int lrow = lane & 15, lhalf = lane >> 4;
  const int kb8 = lhalf * 8, kb16 = lhalf * 16;
  const int wm = wave & 3, wn = wave >> 2;
  const int m0 = blockIdx.y * GBM, n0 = blockIdx.x * GBN;

  // hoist the 4 bias values this lane ever touches
  float bv[4];
  #pragma unroll
  for (int nt = 0; nt < 4; ++nt)
    bv[nt] = bias[n0 + wn * 64 + nt * 16 + lrow];

  v8f acc[2][4];
  #pragma unroll
  for (int i = 0; i < 2; ++i)
    #pragma unroll
    for (int j = 0; j < 4; ++j)
      #pragma unroll
      for (int e = 0; e < 8; ++e) acc[i][j][e] = 0.0f;

  gemm_tiles(A, B, N, K, m0, n0, As, Bst, tid, wm, wn, lrow, kb8, kb16, acc);

  #pragma unroll
  for (int mt = 0; mt < 2; ++mt)
    #pragma unroll
    for (int nt = 0; nt < 4; ++nt) {
      int c = n0 + wn * 64 + nt * 16 + lrow;
      #pragma unroll
      for (int v = 0; v < 8; ++v) {
        int r = m0 + wm * 32 + mt * 16 + v + lhalf * 8;
        C[(size_t)r * N + c] = acc[mt][nt][v] + bv[nt];
      }
    }
}

// ---------------------------------------------------------------- Flash attention
#define SEQ   4096
#define CDIM  512
#define HD    64
#define QKVLD 1536

__global__ __launch_bounds__(128) void flash_attn_wmma(
    const u16b* __restrict__ qkv, u16b* __restrict__ out)
{
  __shared__ alignas(16) u16b Ks[64 * 64];     // [key][d]   row-major
  __shared__ alignas(16) u16b Vt[64 * 64];     // [d][key]   transposed
  __shared__ alignas(16) u16b Ps[4][16 * 64];  // per-wave P strip [row][key]

  const int tid   = threadIdx.x;
  const int wave  = tid >> 5, lane = tid & 31;
  const int lrow  = lane & 15, lhalf = lane >> 4;
  const int kb8   = lhalf * 8, kb16 = lhalf * 16;

  const int bid   = blockIdx.x;
  const int qblk  = bid & 63;
  const int h     = (bid >> 6) & 7;
  const int b     = bid >> 9;
  const int qrow0 = qblk * 64 + wave * 16;

  const u16b* base = qkv + (size_t)b * SEQ * QKVLD + h * HD;

  // Q fragments (A layout): 16 rows x 64 d, two 16x32 chunks
  v16bf qfrag[2];
  #pragma unroll
  for (int kc = 0; kc < 2; ++kc) {
    const u16b* qr = base + (size_t)(qrow0 + lrow) * QKVLD + kc * 32;
    Frag16 f;
    f.lo = *(const uint4*)(qr + kb8);
    f.hi = *(const uint4*)(qr + 16 + kb8);
    qfrag[kc] = frag_as(f);
  }

  float m_i[8], l_i[8];
  v8f o[4];
  #pragma unroll
  for (int v = 0; v < 8; ++v) { m_i[v] = -3.0e38f; l_i[v] = 0.0f; }
  #pragma unroll
  for (int dt = 0; dt < 4; ++dt)
    #pragma unroll
    for (int e = 0; e < 8; ++e) o[dt][e] = 0.0f;

  const int krow = tid >> 1, kcb = (tid & 1) * 32;

  for (int kb = 0; kb < SEQ / 64; ++kb) {
    __syncthreads();
    { // K tile: CDNA5 async global->LDS copy (64B/lane via 4x B128,
      // INST_OFFSET advances both the global and the LDS address).
      unsigned lds_u32 = (unsigned)(uintptr_t)(Ks + krow * 64 + kcb);
      unsigned long long ga = (unsigned long long)(uintptr_t)
          (base + (size_t)(kb * 64 + krow) * QKVLD + CDIM + kcb);
      asm volatile(
          "global_load_async_to_lds_b128 %0, %1, off\n\t"
          "global_load_async_to_lds_b128 %0, %1, off offset:16\n\t"
          "global_load_async_to_lds_b128 %0, %1, off offset:32\n\t"
          "global_load_async_to_lds_b128 %0, %1, off offset:48"
          :: "v"(lds_u32), "v"(ga) : "memory");
    }
    { // V tile transposed: Vt[d][key]
      const uint4* s4 = (const uint4*)(base + (size_t)(kb * 64 + krow) * QKVLD + 2 * CDIM + kcb);
      #pragma unroll
      for (int w = 0; w < 4; ++w) {
        uint4 q = s4[w];
        unsigned vals[4] = {q.x, q.y, q.z, q.w};
        #pragma unroll
        for (int c = 0; c < 4; ++c) {
          int j = kcb + w * 8 + c * 2;
          Vt[(j    ) * 64 + krow] = (u16b)(vals[c] & 0xffffu);
          Vt[(j + 1) * 64 + krow] = (u16b)(vals[c] >> 16);
        }
      }
    }
    if (kb + 1 < SEQ / 64)
      __builtin_prefetch(base + (size_t)((kb + 1) * 64 + krow) * QKVLD + CDIM + kcb, 0, 1);
    asm volatile("s_wait_asynccnt 0x0" ::: "memory");
    __syncthreads();

    // S = Q K^T (4 tiles of 16x16), scale
    float s[4][8];
    #pragma unroll
    for (int nt = 0; nt < 4; ++nt) {
      v8f sa;
      #pragma unroll
      for (int e = 0; e < 8; ++e) sa[e] = 0.0f;
      #pragma unroll
      for (int kc = 0; kc < 2; ++kc) {
        const uint4* p = (const uint4*)(Ks + (nt * 16 + lrow) * 64 + kc * 32);
        Frag16 f; f.lo = p[kb8 >> 3]; f.hi = p[(16 + kb8) >> 3];
        sa = __builtin_amdgcn_wmma_f32_16x16x32_bf16(
            false, qfrag[kc], false, frag_as(f), (short)0, sa, false, false);
      }
      #pragma unroll
      for (int v = 0; v < 8; ++v) s[nt][v] = sa[v] * 0.125f;   // 64^-0.5
    }

    // online softmax: rows live across 16-lane half-groups
    #pragma unroll
    for (int v = 0; v < 8; ++v) {
      float tm = fmaxf(fmaxf(s[0][v], s[1][v]), fmaxf(s[2][v], s[3][v]));
      tm = fmaxf(tm, __shfl_xor(tm, 1, 32));
      tm = fmaxf(tm, __shfl_xor(tm, 2, 32));
      tm = fmaxf(tm, __shfl_xor(tm, 4, 32));
      tm = fmaxf(tm, __shfl_xor(tm, 8, 32));
      float mnew  = fmaxf(m_i[v], tm);
      float alpha = __expf(m_i[v] - mnew);
      m_i[v] = mnew;
      float rs = 0.0f;
      #pragma unroll
      for (int nt = 0; nt < 4; ++nt) {
        float p = __expf(s[nt][v] - mnew);
        s[nt][v] = p; rs += p;
      }
      rs += __shfl_xor(rs, 1, 32);
      rs += __shfl_xor(rs, 2, 32);
      rs += __shfl_xor(rs, 4, 32);
      rs += __shfl_xor(rs, 8, 32);
      l_i[v] = l_i[v] * alpha + rs;
      #pragma unroll
      for (int dt = 0; dt < 4; ++dt) o[dt][v] *= alpha;
    }

    // P (D-layout) -> wave-private LDS strip (A-layout source)
    #pragma unroll
    for (int nt = 0; nt < 4; ++nt)
      #pragma unroll
      for (int v = 0; v < 8; ++v)
        Ps[wave][(v + lhalf * 8) * 64 + nt * 16 + lrow] = to_bf16_bits(s[nt][v]);

    v16bf pfrag[2];
    #pragma unroll
    for (int kc = 0; kc < 2; ++kc) {
      const uint4* p = (const uint4*)(Ps[wave] + lrow * 64 + kc * 32);
      Frag16 f; f.lo = p[kb8 >> 3]; f.hi = p[(16 + kb8) >> 3];
      pfrag[kc] = frag_as(f);
    }
    // O += P V
    #pragma unroll
    for (int dt = 0; dt < 4; ++dt)
      #pragma unroll
      for (int kc = 0; kc < 2; ++kc) {
        const uint4* p = (const uint4*)(Vt + (dt * 16 + lrow) * 64 + kc * 32);
        Frag16 f; f.lo = p[kb8 >> 3]; f.hi = p[(16 + kb8) >> 3];
        o[dt] = __builtin_amdgcn_wmma_f32_16x16x32_bf16(
            false, pfrag[kc], false, frag_as(f), (short)0, o[dt], false, false);
      }
  }

  // normalize + write as bf16 (input to projection GEMM)
  #pragma unroll
  for (int v = 0; v < 8; ++v) {
    float inv = 1.0f / l_i[v];
    int row = qrow0 + v + lhalf * 8;
    #pragma unroll
    for (int dt = 0; dt < 4; ++dt)
      out[((size_t)b * SEQ + row) * CDIM + h * HD + dt * 16 + lrow] =
          to_bf16_bits(o[dt][v] * inv);
  }
}

// ---------------------------------------------------------------- launch
extern "C" void kernel_launch(void* const* d_in, const int* in_sizes, int n_in,
                              void* d_out, int out_size, void* d_ws, size_t ws_size,
                              hipStream_t stream) {
  (void)in_sizes; (void)n_in; (void)out_size; (void)ws_size;
  const float* x      = (const float*)d_in[0];   // [2,4096,512]
  const float* W_qkv  = (const float*)d_in[1];   // [512,1536]
  const float* W_proj = (const float*)d_in[2];   // [512,512]
  const float* b_proj = (const float*)d_in[3];   // [512]
  float* outp = (float*)d_out;

  u16b* ws = (u16b*)d_ws;
  size_t off = 0;
  u16b* xb     = ws + off; off += (size_t)8192 * 512;
  u16b* wqkvb  = ws + off; off += (size_t)512 * 1536;
  u16b* wprojb = ws + off; off += (size_t)512 * 512;
  u16b* qkvb   = ws + off; off += (size_t)8192 * 1536;
  u16b* attnb  = ws + off; off += (size_t)8192 * 512;

  int n4;
  n4 = 8192 * 512 / 4;
  cvt_f32_to_bf16<<<(n4 + 255) / 256, 256, 0, stream>>>(x, xb, n4);
  n4 = 512 * 1536 / 4;
  cvt_f32_to_bf16<<<(n4 + 255) / 256, 256, 0, stream>>>(W_qkv, wqkvb, n4);
  n4 = 512 * 512 / 4;
  cvt_f32_to_bf16<<<(n4 + 255) / 256, 256, 0, stream>>>(W_proj, wprojb, n4);

  // QKV GEMM: [8192,512] x [512,1536] -> bf16 qkv
  gemm_bf16_out<<<dim3(1536 / GBN, 8192 / GBM), 256, 0, stream>>>(
      xb, wqkvb, 8192, 1536, 512, qkvb);

  // Flash attention: 2 batches * 8 heads * 64 query blocks
  flash_attn_wmma<<<dim3(2 * 8 * 64), 128, 0, stream>>>(qkvb, attnb);

  // Projection GEMM + bias: [8192,512] x [512,512] -> f32 out
  gemm_f32_bias_out<<<dim3(512 / GBN, 8192 / GBM), 256, 0, stream>>>(
      attnb, wprojb, 8192, 512, 512, outp, b_proj);
}
